// GATConv_45208825757747
// MI455X (gfx1250) — matrix operand order repaired and checked by reading, not compiled, not run on hardware
//
#include <hip/hip_runtime.h>

typedef __attribute__((ext_vector_type(16))) __bf16 v16bf;
typedef __attribute__((ext_vector_type(8)))  __bf16 v8bf;
typedef __attribute__((ext_vector_type(8)))  float  v8f;

#define B_DIM   8
#define N_NODES 2048
#define IN_DIM  256
#define D_HEAD  64
#define N_HEADS 4
#define NEG_SLOPE 0.2f
#define LDS_STRIDE 40   // padded row stride (elements) for the transposed tile

// ---------------------------------------------------------------------------
// Kernel 1: Wh = h @ W  (M=B*N=16384, K=256, Ncol=256), bf16 WMMA, fp32 acc.
// Output: bf16, transposed per head:  WhT[(b*H+h)][d][n]  ([bh][64][2048]).
// D-fragment rows are consecutive nodes -> one packed 16B bf16 store per lane.
// ---------------------------------------------------------------------------
__global__ __launch_bounds__(256) void gat_gemm_wh(
    const float* __restrict__ h, const float* __restrict__ W,
    __bf16* __restrict__ WhT)
{
    const int wave  = threadIdx.x >> 5;
    const int lane  = threadIdx.x & 31;
    const int mLane = lane & 15;
    const int half  = lane >> 4;

    const int mBase = blockIdx.x * 32 + (wave >> 2) * 16;  // gridDim.x = 512
    const int nBase = blockIdx.y * 64 + (wave & 3) * 16;   // gridDim.y = 4

    v8f acc = {};
    const float* arow = h + (size_t)(mBase + mLane) * IN_DIM;

    for (int k0 = 0; k0 < IN_DIM; k0 += 32) {
        v16bf afrag, bfrag;
        #pragma unroll
        for (int v = 0; v < 8; ++v) {
            const int kk = (v < 4) ? (v * 2 + half * 8) : (16 + (v - 4) * 2 + half * 8);
            afrag[2 * v]     = (__bf16)arow[k0 + kk];
            afrag[2 * v + 1] = (__bf16)arow[k0 + kk + 1];
        }
        #pragma unroll
        for (int v = 0; v < 8; ++v) {
            const int kk = (v < 4) ? (v * 2 + half * 8) : (16 + (v - 4) * 2 + half * 8);
            bfrag[2 * v]     = (__bf16)W[(size_t)(k0 + kk)     * 256 + nBase + mLane];
            bfrag[2 * v + 1] = (__bf16)W[(size_t)(k0 + kk + 1) * 256 + nBase + mLane];
        }
        acc = __builtin_amdgcn_wmma_f32_16x16x32_bf16(
                  false, afrag, false, bfrag, (short)0, acc, false, false);
    }

    // Pack the 8 consecutive-node results of this lane into one 16B store.
    v8bf ov;
    #pragma unroll
    for (int r = 0; r < 8; ++r) ov[r] = (__bf16)acc[r];

    const int gc = nBase + mLane;               // global column = h*64 + d
    const int b  = mBase >> 11;
    const int hh = gc >> 6, d = gc & (D_HEAD - 1);
    const int n0 = (mBase & (N_NODES - 1)) + half * 8;
    *(v8bf*)&WhT[(((size_t)b * N_HEADS + hh) * D_HEAD + d) * N_NODES + n0] = ov;
}

// ---------------------------------------------------------------------------
// Kernel 2: el/er = Wh . a_src / a_dst.  Thread-per-node, fully coalesced
// reads of the transposed bf16 layout.
// ---------------------------------------------------------------------------
__global__ __launch_bounds__(256) void gat_el_er(
    const __bf16* __restrict__ WhT, const float* __restrict__ a_src,
    const float* __restrict__ a_dst, float* __restrict__ el, float* __restrict__ er)
{
    const int bh = blockIdx.y;                       // 0..31
    const int n  = blockIdx.x * 256 + threadIdx.x;   // gridDim.x = 8
    const int hh = bh & (N_HEADS - 1);

    const __bf16* base = WhT + (size_t)bh * D_HEAD * N_NODES + n;
    float v = 0.0f, u = 0.0f;
    #pragma unroll
    for (int d = 0; d < D_HEAD; ++d) {
        const float w = (float)base[(size_t)d * N_NODES];
        v += w * a_src[hh * D_HEAD + d];
        u += w * a_dst[hh * D_HEAD + d];
    }
    el[(size_t)bh * N_NODES + n] = v;
    er[(size_t)bh * N_NODES + n] = u;
}

// ---------------------------------------------------------------------------
// Kernel 3: fused flash-style attention.
//   alpha = softmax_j(lrelu(el_i + er_j), mask),  out = alpha @ Wh + bias
// workgroup = one (b,h) x 128 i-rows; j streamed in 32-steps with
// double-buffered global_load_async_to_lds_b128 staging of WhT tiles.
// ---------------------------------------------------------------------------
__global__ __launch_bounds__(256) void gat_attn(
    const __bf16* __restrict__ WhT, const float* __restrict__ el,
    const float* __restrict__ er, const unsigned char* __restrict__ mask,
    const float* __restrict__ bias, float* __restrict__ out)
{
    __shared__ __bf16 ldsT[2][D_HEAD * LDS_STRIDE];  // 2 x 5 KB transposed tiles
    __shared__ float  ldsEr[N_NODES];                // er with mask baked in (8 KB)

    const int bh    = blockIdx.x;                    // 0..31
    const int b     = bh >> 2, hd = bh & 3;
    const int wave  = threadIdx.x >> 5;
    const int lane  = threadIdx.x & 31;
    const int mLane = lane & 15;
    const int half  = lane >> 4;
    const int iBase = blockIdx.y * 128 + wave * 16;

    const __bf16* WhBH = WhT + (size_t)bh * D_HEAD * N_NODES;
    const int dRow = threadIdx.x >> 2;               // 0..63: tile row this thread DMAs
    const int dChk = threadIdx.x & 3;                // 16B chunk within the row

    // async-prefetch tile 0 (32x64 bf16 = 4KB = 256 lanes x 16B)
    {
        const __bf16* g = WhBH + (size_t)dRow * N_NODES + dChk * 8;
        const unsigned lo = (unsigned)(uintptr_t)&ldsT[0][dRow * LDS_STRIDE + dChk * 8];
        asm volatile("global_load_async_to_lds_b128 %0, %1, off" :: "v"(lo), "v"(g) : "memory");
    }

    // stage er with mask folded in: masked j underflow to weight 0 in softmax
    {
        const float* erBH = er + (size_t)bh * N_NODES;
        const unsigned char* mB = mask + (size_t)b * N_NODES;
        for (int j = threadIdx.x; j < N_NODES; j += 256)
            ldsEr[j] = mB[j] ? erBH[j] : -1.0e9f;
    }

    const float eli = el[(size_t)bh * N_NODES + iBase + mLane];
    float mrow = -3.0e38f, lrow = 0.0f;
    v8f acc[4] = {};
    int buf = 0;

    for (int jBase = 0; jBase < N_NODES; jBase += 32) {
        asm volatile("s_wait_asynccnt 0x0" ::: "memory");
        __syncthreads();    // tile[buf] fully in LDS; tile[buf^1] fully consumed

        if (jBase + 32 < N_NODES) {  // kick off next tile's DMA behind the compute
            const __bf16* g = WhBH + (size_t)dRow * N_NODES + (jBase + 32) + dChk * 8;
            const unsigned lo =
                (unsigned)(uintptr_t)&ldsT[buf ^ 1][dRow * LDS_STRIDE + dChk * 8];
            asm volatile("global_load_async_to_lds_b128 %0, %1, off" :: "v"(lo), "v"(g) : "memory");
        }

        // ---- per-lane scores, matching the A-frag K layout ----
        const float4* p0 = (const float4*)&ldsEr[jBase + half * 16];
        const float4* p1 = (const float4*)&ldsEr[jBase + half * 16 + 8];
        const float4 ea = p0[0], eb = p0[1], ec = p1[0], ed = p1[1];
        const float eArr[16] = {ea.x, ea.y, ea.z, ea.w, eb.x, eb.y, eb.z, eb.w,
                                ec.x, ec.y, ec.z, ec.w, ed.x, ed.y, ed.z, ed.w};

        float s[16], smax = -3.0e38f;
        #pragma unroll
        for (int t = 0; t < 16; ++t) {
            float x = eli + eArr[t];
            x = (x > 0.0f) ? x : NEG_SLOPE * x;
            s[t] = x;
            smax = fmaxf(smax, x);
        }
        smax = fmaxf(smax, __shfl_xor(smax, 16));     // combine halves of row mLane
        const float mnew  = fmaxf(mrow, smax);
        const float rescl = __expf(mrow - mnew);
        mrow = mnew;

        v16bf afrag;
        float ssum = 0.0f;
        #pragma unroll
        for (int t = 0; t < 16; ++t) {
            const float p = __expf(s[t] - mnew);
            ssum += p;
            afrag[t] = (__bf16)p;
        }
        ssum += __shfl_xor(ssum, 16);
        lrow = lrow * rescl + ssum;

        // ---- rescale accumulators (acc VGPR r holds row half*8+r) ----
        float rs[8];
        #pragma unroll
        for (int r = 0; r < 8; ++r) rs[r] = __shfl(rescl, half * 8 + r);
        #pragma unroll
        for (int t = 0; t < 4; ++t)
            #pragma unroll
            for (int r = 0; r < 8; ++r) acc[t][r] *= rs[r];

        // ---- 4 WMMAs: P[16x32] @ Wh[32x64]; B-frag = 2x ds_load_b128 ----
        #pragma unroll
        for (int t = 0; t < 4; ++t) {
            const int n = t * 16 + mLane;
            const v8bf* bp = (const v8bf*)&ldsT[buf][n * LDS_STRIDE + half * 16];
            const v8bf blo = bp[0], bhi = bp[1];
            const v16bf bfrag = __builtin_shufflevector(
                blo, bhi, 0, 1, 2, 3, 4, 5, 6, 7, 8, 9, 10, 11, 12, 13, 14, 15);
            acc[t] = __builtin_amdgcn_wmma_f32_16x16x32_bf16(
                         false, afrag, false, bfrag, (short)0, acc[t], false, false);
        }
        buf ^= 1;
    }

    // ---- finalize: divide by row sums, add bias, write [B,N,H*D] ----
    const float rinv = 1.0f / lrow;
    float rr[8];
    #pragma unroll
    for (int r = 0; r < 8; ++r) rr[r] = __shfl(rinv, half * 8 + r);
    #pragma unroll
    for (int t = 0; t < 4; ++t) {
        const int d  = t * 16 + mLane;
        const float bv = bias[hd * D_HEAD + d];
        #pragma unroll
        for (int r = 0; r < 8; ++r) {
            const int row = iBase + half * 8 + r;
            out[((size_t)b * N_NODES + row) * (N_HEADS * D_HEAD) + hd * D_HEAD + d] =
                acc[t][r] * rr[r] + bv;
        }
    }
}

// ---------------------------------------------------------------------------
extern "C" void kernel_launch(void* const* d_in, const int* in_sizes, int n_in,
                              void* d_out, int out_size, void* d_ws, size_t ws_size,
                              hipStream_t stream) {
    const float*         h     = (const float*)d_in[0];
    const unsigned char* mask  = (const unsigned char*)d_in[1];
    const float*         W     = (const float*)d_in[2];
    const float*         a_src = (const float*)d_in[3];
    const float*         a_dst = (const float*)d_in[4];
    const float*         bias  = (const float*)d_in[5];
    float*               out   = (float*)d_out;

    __bf16* WhT = (__bf16*)d_ws;                                      // 8 MB
    float*  el  = (float*)(WhT + (size_t)B_DIM * N_HEADS * D_HEAD * N_NODES);
    float*  er  = el + (size_t)B_DIM * N_HEADS * N_NODES;             // 256 KB each

    gat_gemm_wh<<<dim3(512, 4), 256, 0, stream>>>(h, W, WhT);
    gat_el_er  <<<dim3(8, 32), 256, 0, stream>>>(WhT, a_src, a_dst, el, er);
    gat_attn   <<<dim3(32, 16), 256, 0, stream>>>(WhT, el, er, mask, bias, out);
}
